// ESN_15951508538265
// MI455X (gfx1250) — matrix operand order repaired and checked by reading, not compiled, not run on hardware
//
#include <hip/hip_runtime.h>
#include <hip/hip_bf16.h>

// ---------------------------------------------------------------------------
// ESN: x_t = tanh(u_t @ w_in^T + diag(w_hh)*x_{t-1} + bias), leakage = 1.
// (1) big bf16 WMMA GEMM P = A @ W^T (time-independent), double-buffered LDS
//     with gfx1250 async global->LDS copies; (2) elementwise tanh scan over L.
// B=16, D=2048, L=512.  GEMM: M=8192, N=2048, K=2048 (~68.7 GFLOP).
// Wave tile 64x32 (4x2 WMMA subtiles) keeps VGPR use < 256 so codegen stays
// free of s_set_vgpr_msb / v_mov shuffles (seen when 4x4 tiles were used).
// ---------------------------------------------------------------------------

typedef __attribute__((ext_vector_type(16))) __bf16       v16bf;
typedef __attribute__((ext_vector_type(8)))  float        v8f;
typedef __attribute__((ext_vector_type(4))) unsigned int  v4u;
typedef __attribute__((ext_vector_type(4)))  int          v4i;
typedef __attribute__((ext_vector_type(2))) unsigned int  v2u;
typedef __attribute__((ext_vector_type(4)))  float        v4f;

#define D_DIM 2048
#define L_DIM 512
#define B_DIM 16
#define M_DIM (B_DIM * L_DIM)   // 8192

#define BLK_M 128
#define BLK_N 128
#define TILE_K 32
#define NK    (D_DIM / TILE_K)  // 64
#define LDSP  40                // bf16 elems per LDS row: 32 + 8 pad -> 80B (16B aligned)

#if defined(__gfx1250__) && __has_builtin(__builtin_amdgcn_global_load_async_to_lds_b128)
#define HAVE_ASYNC 1
#else
#define HAVE_ASYNC 0
#endif

union FragU {
    v16bf v;
    v4u   q[2];
};

__device__ __forceinline__ unsigned short f2bf(float f) {
    unsigned int u = __float_as_uint(f);
    unsigned int r = (u + 0x7FFFu + ((u >> 16) & 1u)) >> 16;  // RNE
    return (unsigned short)r;
}

#if HAVE_ASYNC
__device__ __forceinline__ void async_cp16(const unsigned short* g, unsigned short* l) {
    __builtin_amdgcn_global_load_async_to_lds_b128(
        (__attribute__((address_space(1))) v4i*)g,
        (__attribute__((address_space(3))) v4i*)l,
        0, 0);
}
template <int N>
__device__ __forceinline__ void wait_async() {
#if __has_builtin(__builtin_amdgcn_s_wait_asynccnt)
    __builtin_amdgcn_s_wait_asynccnt(N);
#else
    asm volatile("s_wait_asynccnt %0" ::"n"(N) : "memory");
#endif
}
#endif

// --- Kernel 0: convert w_in (fp32 [p][h]) -> bf16, same layout ([N][K]) -----
__global__ __launch_bounds__(256) void cvt_w_kernel(const float* __restrict__ w,
                                                    unsigned short* __restrict__ wb) {
    int i = (blockIdx.x * 256 + threadIdx.x) * 4;
    v4f v = *(const v4f*)(w + i);
    v2u o;
    o.x = (unsigned int)f2bf(v.x) | ((unsigned int)f2bf(v.y) << 16);
    o.y = (unsigned int)f2bf(v.z) | ((unsigned int)f2bf(v.w) << 16);
    *(v2u*)(wb + i) = o;
}

// --- Kernel 1: build A bf16 [(b,l), h] from u fp32 [b][h][l] (transpose) ----
__global__ __launch_bounds__(256) void build_a_kernel(const float* __restrict__ u,
                                                      unsigned short* __restrict__ A) {
    __shared__ float tile[32][33];
    const int t  = threadIdx.x;
    const int l0 = blockIdx.x * 32;
    const int h0 = blockIdx.y * 32;
    const int b  = blockIdx.z;
    const size_t ubase = (size_t)b * D_DIM * L_DIM;
    {
        int hh = t >> 3;
        int lq = (t & 7) * 4;
        v4f v = *(const v4f*)(u + ubase + (size_t)(h0 + hh) * L_DIM + l0 + lq);
        tile[hh][lq + 0] = v.x;
        tile[hh][lq + 1] = v.y;
        tile[hh][lq + 2] = v.z;
        tile[hh][lq + 3] = v.w;
    }
    __syncthreads();
    {
        int lr = t >> 3;
        int hq = (t & 7) * 4;
        v2u o;
        o.x = (unsigned int)f2bf(tile[hq + 0][lr]) | ((unsigned int)f2bf(tile[hq + 1][lr]) << 16);
        o.y = (unsigned int)f2bf(tile[hq + 2][lr]) | ((unsigned int)f2bf(tile[hq + 3][lr]) << 16);
        *(v2u*)(&A[((size_t)b * L_DIM + l0 + lr) * D_DIM + h0 + hq]) = o;
    }
}

// --- Kernel 2: GEMM P[M,N] = A[M,K] x W[N,K]^T via v_wmma_f32_16x16x32_bf16 -
// 256 threads = 8 waves (2x4); wave tile 64x32 = 4x2 WMMA subtiles.
// Double-buffered LDS; async global->LDS staging.
__global__ __launch_bounds__(256) void gemm_kernel(const unsigned short* __restrict__ A,
                                                   const unsigned short* __restrict__ W,
                                                   float* __restrict__ P) {
    __shared__ __align__(16) unsigned short Alds[2][BLK_M * LDSP];  // 2 x 10 KB
    __shared__ __align__(16) unsigned short Blds[2][BLK_N * LDSP];  // 2 x 10 KB

    const int t    = threadIdx.x;
    const int lane = t & 31;
    const int wave = t >> 5;
    const int wm   = wave >> 2;          // 0..1 : 64 M-rows each
    const int wn   = wave & 3;           // 0..3 : 32 N-cols each
    const int m0   = blockIdx.y * BLK_M;
    const int n0   = blockIdx.x * BLK_N;

    const int l15  = lane & 15;
    const int hiL  = (lane >> 4) & 1;

    // staging: A tile 128x32 and B tile 128x32, 2 x b128 per thread each
    const int row  = t >> 1;
    const int half = t & 1;
    const unsigned short* agp = A + (size_t)(m0 + row) * D_DIM + half * 16;
    const unsigned short* bgp = W + (size_t)(n0 + row) * D_DIM + half * 16;
    const int loff = row * LDSP + half * 16;

    v8f acc[4][2];
#pragma unroll
    for (int i = 0; i < 4; ++i)
#pragma unroll
        for (int j = 0; j < 2; ++j) acc[i][j] = {};

    auto stage = [&](int k0, int buf) {
        const unsigned short* ag = agp + k0;
        const unsigned short* bg = bgp + k0;
        unsigned short*       al = &Alds[buf][loff];
        unsigned short*       bl = &Blds[buf][loff];
#if HAVE_ASYNC
        async_cp16(ag, al);
        async_cp16(ag + 8, al + 8);
        async_cp16(bg, bl);
        async_cp16(bg + 8, bl + 8);
#else
        v4u a0 = *(const v4u*)(ag);
        v4u a1 = *(const v4u*)(ag + 8);
        v4u b0 = *(const v4u*)(bg);
        v4u b1 = *(const v4u*)(bg + 8);
        *(v4u*)(al)     = a0;
        *(v4u*)(al + 8) = a1;
        *(v4u*)(bl)     = b0;
        *(v4u*)(bl + 8) = b1;
#endif
    };

    stage(0, 0);

    for (int kt = 0; kt < NK; ++kt) {
        const int cur = kt & 1;
        if (kt + 1 < NK) {
            stage((kt + 1) * TILE_K, cur ^ 1);
#if HAVE_ASYNC
            wait_async<4>();   // previous tile's 4 transfers retired (in-order)
#endif
        }
#if HAVE_ASYNC
        else {
            wait_async<0>();
        }
#endif
        __syncthreads();       // tile kt visible to all waves

        // A fragments: lanes 0-15 hold K{0..7,16..23}; lanes 16-31 K{8..15,24..31}
        FragU fa[4];
#pragma unroll
        for (int i = 0; i < 4; ++i) {
            const unsigned short* p = &Alds[cur][(wm * 64 + i * 16 + l15) * LDSP + hiL * 8];
            fa[i].q[0] = *(const v4u*)(p);
            fa[i].q[1] = *(const v4u*)(p + 16);
        }
#pragma unroll
        for (int j = 0; j < 2; ++j) {
            // B fragment: lanes 0-15 hold K{0..7,16..23}; lanes 16-31 K{8..15,24..31} (per column n)
            FragU fb;
            const unsigned short* p = &Blds[cur][(wn * 32 + j * 16 + l15) * LDSP + hiL * 8];
            fb.q[0] = *(const v4u*)(p);
            fb.q[1] = *(const v4u*)(p + 16);
#pragma unroll
            for (int i = 0; i < 4; ++i)
                acc[i][j] = __builtin_amdgcn_wmma_f32_16x16x32_bf16(
                    false, fa[i].v, false, fb.v,
                    (short)0, acc[i][j], false, false);
        }
        __syncthreads();       // all waves done with buf[cur] before it is restaged
    }

    // epilogue: C layout -> lanes 0-15: M=r, N=lane ; lanes 16-31: M=r+8, N=lane-16
#pragma unroll
    for (int i = 0; i < 4; ++i)
#pragma unroll
        for (int j = 0; j < 2; ++j) {
            const int mb = m0 + wm * 64 + i * 16 + hiL * 8;
            const int nc = n0 + wn * 32 + j * 16 + l15;
#pragma unroll
            for (int r = 0; r < 8; ++r)
                P[(size_t)(mb + r) * D_DIM + nc] = acc[i][j][r];
        }
}

// --- Kernel 3: leaky-tanh scan over L, LDS-transposed coalesced stores ------
__global__ __launch_bounds__(256) void scan_kernel(const float* __restrict__ P,
                                                   const float* __restrict__ w_hh,
                                                   const float* __restrict__ bias,
                                                   float* __restrict__ out) {
    __shared__ float xs[256][33];
    const int t  = threadIdx.x;
    const int b  = blockIdx.y;
    const int p0 = blockIdx.x * 256;
    const int p  = p0 + t;
    const float wd = w_hh[(size_t)p * D_DIM + p];   // diagonal element
    const float bi = bias[p];
    const float lr = 1.0f;                          // LEAKAGE_RATE
    float x = 0.0f;
    for (int l0 = 0; l0 < L_DIM; l0 += 32) {
#pragma unroll
        for (int j = 0; j < 32; ++j) {
            float pre = P[(size_t)(b * L_DIM + l0 + j) * D_DIM + p] + wd * x + bi;
            x = (1.0f - lr) * x + lr * tanhf(pre);
            xs[t][j] = x;
        }
        __syncthreads();
#pragma unroll
        for (int g = 0; g < 32; ++g) {
            int e  = g * 256 + t;
            int pr = e >> 5;
            int lc = e & 31;
            out[((size_t)b * D_DIM + p0 + pr) * L_DIM + l0 + lc] = xs[pr][lc];
        }
        __syncthreads();
    }
}

// ---------------------------------------------------------------------------
extern "C" void kernel_launch(void* const* d_in, const int* in_sizes, int n_in,
                              void* d_out, int out_size, void* d_ws, size_t ws_size,
                              hipStream_t stream) {
    const float* u    = (const float*)d_in[0];  // [B, D, L]
    const float* w_in = (const float*)d_in[1];  // [D, D]
    const float* w_hh = (const float*)d_in[2];  // [D, D] (diag used)
    const float* bias = (const float*)d_in[3];  // [D]
    float* out = (float*)d_out;                 // [B, D, L]

    const size_t WB_BYTES = (size_t)D_DIM * D_DIM * sizeof(unsigned short); // 8 MB
    const size_t AB_BYTES = (size_t)M_DIM * D_DIM * sizeof(unsigned short); // 32 MB
    unsigned short* Wb = (unsigned short*)d_ws;
    unsigned short* Ab = (unsigned short*)((char*)d_ws + WB_BYTES);
    float*          P  = (float*)((char*)d_ws + WB_BYTES + AB_BYTES);       // 64 MB

    cvt_w_kernel<<<(D_DIM * D_DIM) / (256 * 4), 256, 0, stream>>>(w_in, Wb);
    build_a_kernel<<<dim3(L_DIM / 32, D_DIM / 32, B_DIM), 256, 0, stream>>>(u, Ab);
    gemm_kernel<<<dim3(D_DIM / BLK_N, M_DIM / BLK_M), 256, 0, stream>>>(Ab, Wb, P);
    scan_kernel<<<dim3(D_DIM / 256, B_DIM), 256, 0, stream>>>(P, w_hh, bias, out);
}
